// Attention_30666066493686
// MI455X (gfx1250) — matrix operand order repaired and checked by reading, not compiled, not run on hardware
//
#include <hip/hip_runtime.h>

// Problem constants: T,N,B,D,A = 32,36,64,512,128
constexpr int T_ = 32;
constexpr int N_ = 36;
constexpr int B_ = 64;
constexpr int D_ = 512;
constexpr int A_ = 128;

typedef _Float16 half_t;
typedef __attribute__((ext_vector_type(16))) _Float16    v16h;
typedef __attribute__((ext_vector_type(2)))  float       v2f;
typedef __attribute__((ext_vector_type(4)))  float       v4f;
typedef __attribute__((ext_vector_type(8)))  float       v8f;
typedef __attribute__((ext_vector_type(2)))  unsigned int v2u;
typedef __attribute__((ext_vector_type(4)))  unsigned int v4u;
typedef __attribute__((ext_vector_type(8)))  unsigned int v8u;

// LDS layout (f16 hi/lo planes of the region slice)
constexpr int HS    = 520;   // padded row stride in halves (1040 B, 16B-aligned chunks)
constexpr int NROWS = 37;    // 36 region rows + 1 zero row (M padded to 48)
constexpr int SMEM_BYTES = (2 * NROWS * HS + 2 * D_) * 2 + (48 + 40) * 4;  // ~79.4 KB

// W_att pre-split fragment blob in d_ws: hi[32 steps][8 ntiles][32 lanes][8 dw] then lo
constexpr int    WS_FRAG_DW = 32 * 8 * 32 * 8;             // 65536 dwords per plane
constexpr size_t WS_BYTES   = (size_t)WS_FRAG_DW * 2 * 4;  // 512 KB

__device__ __forceinline__ unsigned int packh(half_t a, half_t b) {
  return (unsigned int)__builtin_bit_cast(unsigned short, a) |
         ((unsigned int)__builtin_bit_cast(unsigned short, b) << 16);
}
__device__ __forceinline__ float h2f_lo(unsigned int u) {
  return (float)__builtin_bit_cast(half_t, (unsigned short)(u & 0xFFFFu));
}
__device__ __forceinline__ float h2f_hi(unsigned int u) {
  return (float)__builtin_bit_cast(half_t, (unsigned short)(u >> 16));
}

__device__ __forceinline__ v8f wmma16(v16h a, v16h b, v8f c) {
  // V_WMMA_F32_16X16X32_F16 : D = A(16x32) * B(32x16) + C, fp32 accumulate
  return __builtin_amdgcn_wmma_f32_16x16x32_f16(false, a, false, b, (short)0, c, false, false);
}

// A-fragment (f16 16x32): lanes 0-15 hold K=[0..15], lanes 16-31 K=[16..31]
__device__ __forceinline__ v16h load_afrag(const half_t* rowBase, int k0, int hi8) {
  v4u c1 = *(const v4u*)(rowBase + k0 + 2 * hi8);        // ds_load_b128
  v4u c2 = *(const v4u*)(rowBase + k0 + 2 * hi8 + 8);   // ds_load_b128
  v8u u;
  u[0] = c1[0]; u[1] = c1[1]; u[2] = c1[2]; u[3] = c1[3];
  u[4] = c2[0]; u[5] = c2[1]; u[6] = c2[2]; u[7] = c2[3];
  return __builtin_bit_cast(v16h, u);
}

// cf is the row-constant frame-projection tile; fold per-element before tanh.
__device__ __forceinline__ void epilogue_tile(v8f c, v8f cf, int mtBase, float ba, float wf,
                                              int nn, int hi, float* sAtt) {
#pragma unroll
  for (int r = 0; r < 8; ++r) {
    // C layout: VGPR r, lane l -> (M = mtBase + r + 8*(l>>4), N = n0 + (l&15))
    float h = tanhf(c[r] + cf[r] + ba) * wf;
    h += __shfl_xor(h, 1);
    h += __shfl_xor(h, 2);
    h += __shfl_xor(h, 4);
    h += __shfl_xor(h, 8);
    if (nn == 0) atomicAdd(&sAtt[mtBase + r + 8 * hi], h);   // ds_add_f32
  }
}

// Prep: split W_att (1024x128 fp32) into f16 hi/lo planes, stored in B-fragment order.
// B-fragment (f16 32x16): lane l holds K=[kbase..kbase+15] of column n; dword v packs K=2v,2v+1
__global__ __launch_bounds__(256) void prep_watt(const float* __restrict__ Watt,
                                                 unsigned int* __restrict__ ws) {
  int idx = blockIdx.x * 256 + threadIdx.x;     // 32 steps * 8 ntiles * 32 lanes = 8192
  if (idx >= 32 * 8 * 32) return;
  int lane = idx & 31;
  int j    = (idx >> 5) & 7;
  int s    = idx >> 8;
  int kbase = s * 32 + (lane >> 4) * 16;
  int n     = j * 16 + (lane & 15);
  unsigned int* hi = ws + ((size_t)(s * 8 + j) * 32 + lane) * 8;
  unsigned int* lo = hi + WS_FRAG_DW;
#pragma unroll
  for (int v = 0; v < 8; ++v) {
    float w0 = Watt[(size_t)(kbase + 2 * v) * A_ + n];
    float w1 = Watt[(size_t)(kbase + 2 * v + 1) * A_ + n];
    half_t h0 = (half_t)w0, h1 = (half_t)w1;
    half_t l0 = (half_t)(w0 - (float)h0), l1 = (half_t)(w1 - (float)h1);
    hi[v] = packh(h0, h1);
    lo[v] = packh(l0, l1);
  }
}

template <bool WSB>
__global__ __launch_bounds__(256) void att_fused(
    const float* __restrict__ region,        // [T,N,B,D]
    const float* __restrict__ frame,         // [T,B,D]
    const unsigned char* __restrict__ mask,  // [T,N,B] bool (1 byte)
    const float* __restrict__ Watt,          // [1024,128]
    const float* __restrict__ batt,          // [A]
    const float* __restrict__ Wfull,         // [A]
    const float* __restrict__ bfull,         // [1]
    const unsigned int* __restrict__ wsB,    // pre-split W_att fragments (WSB path)
    float* __restrict__ out)                 // [T,B,D]
{
  extern __shared__ __align__(16) unsigned char smemRaw[];
  half_t* hiA  = (half_t*)smemRaw;           // NROWS*HS halves (row 36 = zeros)
  half_t* loA  = hiA + NROWS * HS;
  half_t* fHi  = loA + NROWS * HS;           // D_ halves
  half_t* fLo  = fHi + D_;
  float*  sAtt   = (float*)(fLo + D_);       // 48 logits
  float*  sAlpha = sAtt + 48;                // 40

  const int tb   = blockIdx.x;               // one workgroup per (t,b)
  const int t    = tb / B_;
  const int b    = tb % B_;
  const int tid  = threadIdx.x;
  const int wave = tid >> 5;                 // 0..7 (wave32)
  const int lane = tid & 31;

  // ------- Phase A: stage region slice + frame row as f16 hi/lo planes ------
  {
    const float* rbase = region + ((size_t)t * N_) * (size_t)(B_ * D_) + (size_t)b * D_;
    for (int i = tid; i < N_ * (D_ / 4); i += 256) {   // 4608 = 18*256, uniform
      int n  = i / (D_ / 4);
      int c4 = i % (D_ / 4);
      v4f v = *(const v4f*)(rbase + (size_t)n * (B_ * D_) + c4 * 4);
      half_t h0 = (half_t)v[0], h1 = (half_t)v[1], h2 = (half_t)v[2], h3 = (half_t)v[3];
      v2u hw; hw[0] = packh(h0, h1); hw[1] = packh(h2, h3);
      v2u lw; lw[0] = packh((half_t)(v[0] - (float)h0), (half_t)(v[1] - (float)h1));
              lw[1] = packh((half_t)(v[2] - (float)h2), (half_t)(v[3] - (float)h3));
      *(v2u*)(hiA + n * HS + c4 * 4) = hw;   // 8B aligned
      *(v2u*)(loA + n * HS + c4 * 4) = lw;
    }
    // zero row (padded M rows 36..47 read this)
    for (int i = tid; i < HS / 2; i += 256) {
      ((unsigned int*)(hiA + 36 * HS))[i] = 0u;
      ((unsigned int*)(loA + 36 * HS))[i] = 0u;
    }
    const float* fbase = frame + ((size_t)t * B_ + b) * (size_t)D_;
    for (int i = tid; i < D_ / 4; i += 256) {
      v4f v = *(const v4f*)(fbase + i * 4);
      half_t h0 = (half_t)v[0], h1 = (half_t)v[1], h2 = (half_t)v[2], h3 = (half_t)v[3];
      v2u hw; hw[0] = packh(h0, h1); hw[1] = packh(h2, h3);
      v2u lw; lw[0] = packh((half_t)(v[0] - (float)h0), (half_t)(v[1] - (float)h1));
              lw[1] = packh((half_t)(v[2] - (float)h2), (half_t)(v[3] - (float)h3));
      *(v2u*)(fHi + i * 4) = hw;
      *(v2u*)(fLo + i * 4) = lw;
    }
    if (tid < 48) sAtt[tid] = bfull[0];      // att starts at b_full
  }
  __syncthreads();

  // ------- Phase B: (48 x 1024) @ (1024 x 128) via split-f16 WMMA -----------
  const int n0  = wave * 16;
  const int nn  = lane & 15;
  const int hiL = lane >> 4;
  const int hi8 = hiL * 8;                   // A chunk sub-offset (halves)

  int r2 = 32 + nn; if (r2 >= N_) r2 = 36;   // padded rows -> zero row
  const half_t* r0h = hiA + (0 + nn) * HS;   const half_t* r0l = loA + (0 + nn) * HS;
  const half_t* r1h = hiA + (16 + nn) * HS;  const half_t* r1l = loA + (16 + nn) * HS;
  const half_t* r2h = hiA + r2 * HS;         const half_t* r2l = loA + r2 * HS;
  const float*  Bcol = Watt + n0 + nn;       // fallback path column base

  v8f c0 = {}; v8f c1 = {}; v8f c2 = {}; v8f cf = {};

  // B-fragment loader for K-step s (0..31)
  auto load_bfrag = [&](int s, v16h& bh, v16h& bl) {
    if constexpr (WSB) {
      const unsigned int* p = wsB + ((size_t)(s * 8 + wave) * 32 + lane) * 8;
      bh = __builtin_bit_cast(v16h, *(const v8u*)p);                 // 32B coalesced
      bl = __builtin_bit_cast(v16h, *(const v8u*)(p + WS_FRAG_DW));
    } else {
      int kb = s * 32 + hiL * 16;
      v8u uh, ul;
#pragma unroll
      for (int v = 0; v < 8; ++v) {
        float w0 = Bcol[(size_t)(kb + 2 * v) * A_];
        float w1 = Bcol[(size_t)(kb + 2 * v + 1) * A_];
        half_t h0 = (half_t)w0, h1 = (half_t)w1;
        uh[v] = packh(h0, h1);
        ul[v] = packh((half_t)(w0 - (float)h0), (half_t)(w1 - (float)h1));
      }
      bh = __builtin_bit_cast(v16h, uh);
      bl = __builtin_bit_cast(v16h, ul);
    }
  };

  // --- region half of K (s = 0..15): 3 M-tiles, 9 WMMAs per step ---
#pragma unroll 2
  for (int s = 0; s < 16; ++s) {
    v16h bh, bl;
    load_bfrag(s, bh, bl);
    const int k0 = 32 * s;
    v16h a0h = load_afrag(r0h, k0, hi8);  v16h a0l = load_afrag(r0l, k0, hi8);
    v16h a1h = load_afrag(r1h, k0, hi8);  v16h a1l = load_afrag(r1l, k0, hi8);
    v16h a2h = load_afrag(r2h, k0, hi8);  v16h a2l = load_afrag(r2l, k0, hi8);
    // split-precision accumulate: Ahi*Bhi + Ahi*Blo + Alo*Bhi
    c0 = wmma16(a0h, bh, c0); c0 = wmma16(a0h, bl, c0); c0 = wmma16(a0l, bh, c0);
    c1 = wmma16(a1h, bh, c1); c1 = wmma16(a1h, bl, c1); c1 = wmma16(a1l, bh, c1);
    c2 = wmma16(a2h, bh, c2); c2 = wmma16(a2h, bl, c2); c2 = wmma16(a2l, bh, c2);
  }

  // --- frame half of K (s = 16..31): row-constant tile, 3 WMMAs per step ---
#pragma unroll 2
  for (int s = 16; s < 32; ++s) {
    v16h bh, bl;
    load_bfrag(s, bh, bl);
    const int k0 = 32 * (s - 16);
    v16h fh = load_afrag(fHi, k0, hi8);    // broadcast reads
    v16h fl = load_afrag(fLo, k0, hi8);
    cf = wmma16(fh, bh, cf); cf = wmma16(fh, bl, cf); cf = wmma16(fl, bh, cf);
  }

  // Epilogue: att[m] += sum_n tanh(c[m][n] + cf[n] + b_att[n]) * W_full[n]
  {
    const float ba = batt[n0 + nn];
    const float wf = Wfull[n0 + nn];
    epilogue_tile(c0, cf,  0, ba, wf, nn, hiL, sAtt);
    epilogue_tile(c1, cf, 16, ba, wf, nn, hiL, sAtt);
    epilogue_tile(c2, cf, 32, ba, wf, nn, hiL, sAtt);
  }
  __syncthreads();

  // ------- Phase C: mask + softmax over N=36 --------------------------------
  if (tid < N_) {
    bool m = mask[((size_t)t * N_ + tid) * B_ + b] != 0;
    sAtt[tid] = m ? -1.0e9f : sAtt[tid];
  }
  __syncthreads();
  if (tid == 0) {
    float mx = sAtt[0];
    for (int n = 1; n < N_; ++n) mx = fmaxf(mx, sAtt[n]);
    float ssum = 0.0f;
    for (int n = 0; n < N_; ++n) { float e = expf(sAtt[n] - mx); sAlpha[n] = e; ssum += e; }
    float inv = 1.0f / ssum;
    for (int n = 0; n < N_; ++n) sAlpha[n] *= inv;
  }
  __syncthreads();

  // ------- Phase D: out[t,b,d] = sum_n (hi+lo)[n][d] * alpha[n] -------------
  {
    const int dw = tid;                      // dword = 2 consecutive d's
    float acc0 = 0.0f, acc1 = 0.0f;
#pragma unroll
    for (int n = 0; n < N_; ++n) {
      unsigned int h = *(const unsigned int*)(hiA + n * HS + 2 * dw);
      unsigned int l = *(const unsigned int*)(loA + n * HS + 2 * dw);
      float a = sAlpha[n];
      acc0 += (h2f_lo(h) + h2f_lo(l)) * a;
      acc1 += (h2f_hi(h) + h2f_hi(l)) * a;
    }
    v2f o; o[0] = acc0; o[1] = acc1;
    *(v2f*)(out + ((size_t)t * B_ + b) * (size_t)D_ + 2 * dw) = o;
  }
}

extern "C" void kernel_launch(void* const* d_in, const int* in_sizes, int n_in,
                              void* d_out, int out_size, void* d_ws, size_t ws_size,
                              hipStream_t stream) {
  (void)in_sizes; (void)n_in; (void)out_size;
  const float*         region = (const float*)d_in[0];
  const float*         frame  = (const float*)d_in[1];
  const unsigned char* mask   = (const unsigned char*)d_in[2];
  const float*         Watt   = (const float*)d_in[3];
  const float*         batt   = (const float*)d_in[4];
  const float*         Wfull  = (const float*)d_in[5];
  const float*         bfull  = (const float*)d_in[6];
  float*               out    = (float*)d_out;

  const size_t smemBytes = SMEM_BYTES;
  if (d_ws != nullptr && ws_size >= WS_BYTES) {
    unsigned int* ws = (unsigned int*)d_ws;
    prep_watt<<<dim3(32), dim3(256), 0, stream>>>(Watt, ws);
    (void)hipFuncSetAttribute((const void*)att_fused<true>,
                              hipFuncAttributeMaxDynamicSharedMemorySize, (int)smemBytes);
    att_fused<true><<<dim3(T_ * B_), dim3(256), smemBytes, stream>>>(
        region, frame, mask, Watt, batt, Wfull, bfull, ws, out);
  } else {
    (void)hipFuncSetAttribute((const void*)att_fused<false>,
                              hipFuncAttributeMaxDynamicSharedMemorySize, (int)smemBytes);
    att_fused<false><<<dim3(T_ * B_), dim3(256), smemBytes, stream>>>(
        region, frame, mask, Watt, batt, Wfull, bfull, nullptr, out);
  }
}
